// BitLinear_45921790329116
// MI455X (gfx1250) — 8-bit matrix operand order repaired and checked by reading, compiled, not run on hardware
//
#include <hip/hip_runtime.h>
#include <hip/hip_bf16.h>


typedef __attribute__((ext_vector_type(8))) int v8i;

union U8 {
    v8i  v;
    int  i[8];
    int2 p[4];
    int4 q[2];
};

#ifndef USE_ASYNC_LDS
#define USE_ASYNC_LDS 1
#endif

__device__ __forceinline__ void cp16_g2l(void* lds, const void* g) {
#if USE_ASYNC_LDS
    unsigned           l  = (unsigned)(unsigned long long)lds;
    unsigned long long ga = (unsigned long long)g;
    asm volatile("global_load_async_to_lds_b128 %0, %1, off"
                 :: "v"(l), "v"(ga) : "memory");
#else
    *(int4*)lds = *(const int4*)g;
#endif
}

template <int N>
__device__ __forceinline__ void wait_async() {
#if USE_ASYNC_LDS
    asm volatile("s_wait_asynccnt %0" :: "i"(N) : "memory");
#endif
}

__global__ void k_absmax(const float* __restrict__ x, float* __restrict__ part, int n) {
    __shared__ float sm[256];
    float m = 0.f;
    for (int i = blockIdx.x * blockDim.x + threadIdx.x; i < n; i += gridDim.x * blockDim.x)
        m = fmaxf(m, fabsf(x[i]));
    sm[threadIdx.x] = m;
    __syncthreads();
    for (int s = 128; s > 0; s >>= 1) {
        if (threadIdx.x < s) sm[threadIdx.x] = fmaxf(sm[threadIdx.x], sm[threadIdx.x + s]);
        __syncthreads();
    }
    if (threadIdx.x == 0) part[blockIdx.x] = sm[0];
}

__global__ void k_abssum(const float* __restrict__ w, float* __restrict__ part, int n) {
    __shared__ float sm[256];
    float s = 0.f;
    for (int i = blockIdx.x * blockDim.x + threadIdx.x; i < n; i += gridDim.x * blockDim.x)
        s += fabsf(w[i]);
    sm[threadIdx.x] = s;
    __syncthreads();
    for (int t = 128; t > 0; t >>= 1) {
        if (threadIdx.x < t) sm[threadIdx.x] += sm[threadIdx.x + t];
        __syncthreads();
    }
    if (threadIdx.x == 0) part[blockIdx.x] = sm[0];
}

__global__ void k_finalize(const float* __restrict__ px, const float* __restrict__ pw,
                           float* __restrict__ scal, int np, float inv_count) {
    __shared__ float sm[256];
    __shared__ float ss[256];
    float m = 0.f, s = 0.f;
    for (int i = threadIdx.x; i < np; i += 256) {
        m = fmaxf(m, px[i]);
        s += pw[i];
    }
    sm[threadIdx.x] = m;
    ss[threadIdx.x] = s;
    __syncthreads();
    for (int t = 128; t > 0; t >>= 1) {
        if (threadIdx.x < t) {
            sm[threadIdx.x] = fmaxf(sm[threadIdx.x], sm[threadIdx.x + t]);
            ss[threadIdx.x] += ss[threadIdx.x + t];
        }
        __syncthreads();
    }
    if (threadIdx.x == 0) {
        float gamma  = fmaxf(sm[0], 1e-8f);
        float gammaw = fmaxf(ss[0] * inv_count, 1e-8f);
        scal[0] = 127.f / gamma;
        scal[1] = (gamma / 127.f) * gammaw;
        scal[2] = 1.f / gammaw;
    }
}

__global__ void k_quant_x(const float* __restrict__ x, const float* __restrict__ scal,
                          signed char* __restrict__ xq, int n4) {
    int i = blockIdx.x * blockDim.x + threadIdx.x;
    if (i >= n4) return;
    float  s = scal[0];
    float4 v = *(const float4*)(x + (size_t)i * 4);
    char4  r;
    r.x = (signed char)fminf(fmaxf(rintf(v.x * s), -127.f), 127.f);
    r.y = (signed char)fminf(fmaxf(rintf(v.y * s), -127.f), 127.f);
    r.z = (signed char)fminf(fmaxf(rintf(v.z * s), -127.f), 127.f);
    r.w = (signed char)fminf(fmaxf(rintf(v.w * s), -127.f), 127.f);
    *(char4*)(xq + (size_t)i * 4) = r;
}

__global__ void k_quant_w(const float* __restrict__ w, const float* __restrict__ scal,
                          signed char* __restrict__ wq, int n4) {
    int i = blockIdx.x * blockDim.x + threadIdx.x;
    if (i >= n4) return;
    float  s = scal[2];
    float4 v = *(const float4*)(w + (size_t)i * 4);
    char4  r;
    r.x = (signed char)fminf(fmaxf(rintf(v.x * s), -1.f), 1.f);
    r.y = (signed char)fminf(fmaxf(rintf(v.y * s), -1.f), 1.f);
    r.z = (signed char)fminf(fmaxf(rintf(v.z * s), -1.f), 1.f);
    r.w = (signed char)fminf(fmaxf(rintf(v.w * s), -1.f), 1.f);
    *(char4*)(wq + (size_t)i * 4) = r;
}

#define BM  128
#define BN  128
#define BK  64
#define LDT 80

__global__ __launch_bounds__(256)
void k_gemm_i8(const signed char* __restrict__ Aq,
               const signed char* __restrict__ Bq,
               const float* __restrict__ bias,
               const float* __restrict__ scal,
               float* __restrict__ out,
               int M, int N, int K) {
    __shared__ signed char sA[2][BM * LDT];
    __shared__ signed char sB[2][BN * LDT];

    const int tid  = threadIdx.x;
    const int lane = tid & 31;
    const int wave = tid >> 5;
    const int wm   = wave & 3;
    const int wn   = wave >> 2;
    const int l16  = lane & 15;
    const int half = lane >> 4;
    const int bm   = blockIdx.y * BM;
    const int bn   = blockIdx.x * BN;

    v8i acc[2][4] = {};

    auto stage = [&](int buf, int kt) {
        const int k0 = kt * BK;
#pragma unroll
        for (int it = 0; it < 2; ++it) {
            int chunk = tid + it * 256;
            int row   = chunk >> 2;
            int koff  = (chunk & 3) << 4;
            cp16_g2l(&sA[buf][row * LDT + koff],
                     Aq + (size_t)(bm + row) * K + (k0 + koff));
            cp16_g2l(&sB[buf][row * LDT + koff],
                     Bq + (size_t)(bn + row) * K + (k0 + koff));
        }
#if !USE_ASYNC_LDS
#endif
    };

    auto compute = [&](int buf) {
        const signed char* As = sA[buf];
        const signed char* Bs = sB[buf];

        U8 afr[2];
#pragma unroll
        for (int ti = 0; ti < 2; ++ti) {
            const signed char* p = As + (wm * 32 + ti * 16 + l16) * LDT + half * 8;
            afr[ti].p[0] = *(const int2*)(p + 0);
            afr[ti].p[1] = *(const int2*)(p + 16);
            afr[ti].p[2] = *(const int2*)(p + 32);
            afr[ti].p[3] = *(const int2*)(p + 48);
        }
        U8 bfr[4];
#pragma unroll
        for (int tj = 0; tj < 4; ++tj) {
            const signed char* p = Bs + (wn * 64 + tj * 16 + l16) * LDT + half * 8;
            bfr[tj].p[0] = *(const int2*)(p + 0);
            bfr[tj].p[1] = *(const int2*)(p + 16);
            bfr[tj].p[2] = *(const int2*)(p + 32);
            bfr[tj].p[3] = *(const int2*)(p + 48);
        }
#pragma unroll
        for (int ti = 0; ti < 2; ++ti)
#pragma unroll
            for (int tj = 0; tj < 4; ++tj)
                acc[ti][tj] = __builtin_amdgcn_wmma_i32_16x16x64_iu8(
                    true, afr[ti].v, true, bfr[tj].v, acc[ti][tj], false, false);
    };

    const int NT = K / BK;
    stage(0, 0);
    for (int kt = 0; kt < NT; ++kt) {
        const int cur = kt & 1;
        if (kt + 1 < NT) {
            stage(cur ^ 1, kt + 1);
            wait_async<4>();
        } else {
            wait_async<0>();
        }
        __syncthreads();
        compute(cur);
        __syncthreads();
    }

    const float so = scal[1];
#pragma unroll
    for (int tj = 0; tj < 4; ++tj) {
        const int   col = bn + wn * 64 + tj * 16 + l16;
        const float bv  = bias[col];
#pragma unroll
        for (int ti = 0; ti < 2; ++ti) {
            U8 c;
            c.v = acc[ti][tj];
#pragma unroll
            for (int j = 0; j < 8; ++j) {
                const int row = bm + wm * 32 + ti * 16 + half * 8 + j;
                out[(size_t)row * N + col] = (float)c.i[j] * so + bv;
            }
        }
    }
}

extern "C" void kernel_launch(void* const* d_in, const int* in_sizes, int n_in,
                              void* d_out, int out_size, void* d_ws, size_t ws_size,
                              hipStream_t stream) {
    const float* x = (const float*)d_in[0];
    const float* W = (const float*)d_in[1];
    const float* b = (const float*)d_in[2];
    float* out = (float*)d_out;

    const int N  = in_sizes[2];
    const int K  = in_sizes[1] / N;
    const int M  = in_sizes[0] / K;
    const int nx = in_sizes[0];
    const int nw = in_sizes[1];

    float*       scal = (float*)d_ws;
    float*       px   = (float*)((char*)d_ws + 256);
    float*       pw   = (float*)((char*)d_ws + 256 + 4096);
    signed char* xq   = (signed char*)d_ws + 16384;
    signed char* wq   = xq + (size_t)M * K;

    k_absmax<<<1024, 256, 0, stream>>>(x, px, nx);
    k_abssum<<<1024, 256, 0, stream>>>(W, pw, nw);
    k_finalize<<<1, 256, 0, stream>>>(px, pw, scal, 1024, 1.0f / (float)nw);

    k_quant_x<<<(nx / 4 + 255) / 256, 256, 0, stream>>>(x, scal, xq, nx / 4);
    k_quant_w<<<(nw / 4 + 255) / 256, 256, 0, stream>>>(W, scal, wq, nw / 4);

    dim3 grid(N / BN, M / BM);
    k_gemm_i8<<<grid, 256, 0, stream>>>(xq, wq, b, scal, out, M, N, K);
}
